// GCN_39058432590432
// MI455X (gfx1250) — compile-verified, matrix operand order checked by reading, not run on hardware
//
#include <hip/hip_runtime.h>
#include <hip/hip_bf16.h>

#define NN       20000
#define NE       320000
#define ET       (NE + NN)          // edges + self loops
#define KIN      256
#define HID      128
#define NH1      5
#define NH2      3
#define NC1      (NH1 * HID)        // 640
#define NC2      (NH2 * HID)        // 384
#define NBB      4096
#define ATT_SLOPE 0.2f
#define ACT_SLOPE 0.01f

typedef __attribute__((ext_vector_type(16))) __bf16 v16bf;
typedef __attribute__((ext_vector_type(8)))  float  v8f;

static __device__ __forceinline__ unsigned short f32_to_bf16(float f) {
  unsigned int u = __builtin_bit_cast(unsigned int, f);
  u += 0x7fffu + ((u >> 16) & 1u);              // round-to-nearest-even
  return (unsigned short)(u >> 16);
}
static __device__ __forceinline__ unsigned int ord_encode(float f) {
  unsigned int u = __builtin_bit_cast(unsigned int, f);
  return (u & 0x80000000u) ? ~u : (u | 0x80000000u);
}
static __device__ __forceinline__ float ord_decode(unsigned int o) {
  unsigned int u = (o & 0x80000000u) ? (o & 0x7fffffffu) : ~o;
  return __builtin_bit_cast(float, u);
}
static __device__ __forceinline__ float leaky(float x, float s) {
  return x >= 0.f ? x : s * x;
}

// ---------------------------------------------------------------- utilities
__global__ void k_zero_u32(unsigned int* __restrict__ p, long n) {
  long i = (long)blockIdx.x * blockDim.x + threadIdx.x;
  if (i < n) p[i] = 0u;
}

__global__ void k_f32_to_bf16(const float* __restrict__ s,
                              unsigned short* __restrict__ d, long n) {
  long i = (long)blockIdx.x * blockDim.x + threadIdx.x;
  if (i < n) d[i] = f32_to_bf16(s[i]);
}

// W: [K,N] row-major f32  ->  Wt: [N,K] bf16
__global__ void k_transpose_bf16(const float* __restrict__ W,
                                 unsigned short* __restrict__ Wt, int K, int N) {
  int k = blockIdx.x * blockDim.x + threadIdx.x;
  int n = blockIdx.y;
  if (k < K) Wt[(long)n * K + k] = f32_to_bf16(W[(long)k * N + n]);
}

// ------------------------------------------------------------- WMMA GEMM
// C[M,N] (f32) = A[M,K] (bf16, row-major) @ Bt[N,K]^T (bf16, row-major-over-K)
// one wave -> one 16x16 tile; 8 waves/block -> 16x128 strip.
// M%16==0, N%128==0, K%32==0 guaranteed by problem dims.
__global__ void k_wmma_gemm(const unsigned short* __restrict__ A,
                            const unsigned short* __restrict__ Bt,
                            float* __restrict__ C, int M, int K, int N) {
  const int lane = threadIdx.x & 31;
  const int wave = threadIdx.x >> 5;
  const int m0 = blockIdx.x * 16;
  const int n0 = (blockIdx.y * 8 + wave) * 16;
  const int hf    = (lane >> 4) & 1;   // lane half selects K sub-block
  const int l15   = lane & 15;
  const int kbase = hf * 8;

  const unsigned short* arow = A  + (long)(m0 + l15) * K;
  const unsigned short* brow = Bt + (long)(n0 + l15) * K;

  union Frag { v16bf v; unsigned int u[8]; };
  v8f acc = {0.f, 0.f, 0.f, 0.f, 0.f, 0.f, 0.f, 0.f};

  for (int k0 = 0; k0 < K; k0 += 32) {
    Frag a, b;
#pragma unroll
    for (int j = 0; j < 8; ++j) {
      // ISA bf16 A/B fragment layout: dword j holds K pair at this offset
      const int koff = (j < 4) ? (kbase + 2 * j) : (16 + kbase + 2 * (j - 4));
      a.u[j] = *(const unsigned int*)(arow + k0 + koff);
      b.u[j] = *(const unsigned int*)(brow + k0 + koff);
    }
    acc = __builtin_amdgcn_wmma_f32_16x16x32_bf16(false, a.v, false, b.v,
                                                  (short)0, acc, false, false);
  }

  // C/D layout: vgpr v -> M = v + 8*half, lane -> N = l15
  float* crow = C + (long)m0 * N + n0 + l15;
#pragma unroll
  for (int v = 0; v < 8; ++v)
    crow[(long)(v + hf * 8) * N] = acc[v];
}

// ----------------------------------------------------- attention scores
// one wave per (node, head):  a_s[n,h] = sum_c hW[n,h*C+c]*att_s[h,c]
__global__ void k_att_scores(const float* __restrict__ hW,
                             const float* __restrict__ att_s,
                             const float* __restrict__ att_d,
                             float* __restrict__ as_, float* __restrict__ ad_,
                             int heads) {
  const int wid  = blockIdx.x * (blockDim.x >> 5) + (threadIdx.x >> 5);
  const int lane = threadIdx.x & 31;
  if (wid >= NN * heads) return;
  const int node = wid / heads;
  const int h    = wid - node * heads;
  const float* base = hW + (long)node * (heads * HID) + h * HID;
  float ss = 0.f, sd = 0.f;
#pragma unroll
  for (int i = 0; i < HID / 32; ++i) {
    const float v = base[lane + 32 * i];
    ss += v * att_s[h * HID + lane + 32 * i];
    sd += v * att_d[h * HID + lane + 32 * i];
  }
#pragma unroll
  for (int m = 16; m; m >>= 1) {
    ss += __shfl_xor(ss, m, 32);
    sd += __shfl_xor(sd, m, 32);
  }
  if (lane == 0) { as_[wid] = ss; ad_[wid] = sd; }
}

static __device__ __forceinline__ void edge_nodes(const int* ei, int e,
                                                  int& s, int& d) {
  if (e < NE) { s = ei[e]; d = ei[NE + e]; }
  else        { s = e - NE; d = s; }        // appended self-loop
}

// pass 1: segment max (order-preserving uint atomicMax)
__global__ void k_edge_max(const int* __restrict__ ei,
                           const float* __restrict__ as_,
                           const float* __restrict__ ad_,
                           unsigned int* __restrict__ smax, int heads) {
  const int e = blockIdx.x * blockDim.x + threadIdx.x;
  const int h = blockIdx.y;
  if (e >= ET) return;
  int s, d; edge_nodes(ei, e, s, d);
  const float sc = leaky(as_[s * heads + h] + ad_[d * heads + h], ATT_SLOPE);
  atomicMax(&smax[d * heads + h], ord_encode(sc));
}

// pass 2: segment sum of exp(e - max)
__global__ void k_edge_sum(const int* __restrict__ ei,
                           const float* __restrict__ as_,
                           const float* __restrict__ ad_,
                           const unsigned int* __restrict__ smax,
                           float* __restrict__ ssum, int heads) {
  const int e = blockIdx.x * blockDim.x + threadIdx.x;
  const int h = blockIdx.y;
  if (e >= ET) return;
  int s, d; edge_nodes(ei, e, s, d);
  const float sc = leaky(as_[s * heads + h] + ad_[d * heads + h], ATT_SLOPE);
  const float m  = ord_decode(smax[d * heads + h]);
  atomicAdd(&ssum[d * heads + h], __expf(sc - m));
}

// pass 3: out[dst] += alpha * h[src]; one wave per (edge, head)
__global__ void k_edge_aggr(const int* __restrict__ ei,
                            const float* __restrict__ as_,
                            const float* __restrict__ ad_,
                            const unsigned int* __restrict__ smax,
                            const float* __restrict__ ssum,
                            const float* __restrict__ hW,
                            float* __restrict__ out, int heads) {
  const int wave = threadIdx.x >> 5, lane = threadIdx.x & 31;
  const int e = blockIdx.x * (blockDim.x >> 5) + wave;
  const int h = blockIdx.y;
  if (e >= ET) return;
  int s, d; edge_nodes(ei, e, s, d);
  const float sc    = leaky(as_[s * heads + h] + ad_[d * heads + h], ATT_SLOPE);
  const float m     = ord_decode(smax[d * heads + h]);
  const float alpha = __expf(sc - m) / (ssum[d * heads + h] + 1e-16f);
  const float* hsrc = hW  + (long)s * (heads * HID) + h * HID;
  float*       odst = out + (long)d * (heads * HID) + h * HID;
#pragma unroll
  for (int i = 0; i < HID / 32; ++i)
    atomicAdd(&odst[lane + 32 * i], alpha * hsrc[lane + 32 * i]);
}

// layer-1 epilogue: bias + leaky + bf16 for GEMM2 input
__global__ void k_post1(const float* __restrict__ out1,
                        const float* __restrict__ b1,
                        unsigned short* __restrict__ h1bf) {
  long i = (long)blockIdx.x * blockDim.x + threadIdx.x;
  if (i >= (long)NN * NC1) return;
  const int j = (int)(i % NC1);
  h1bf[i] = f32_to_bf16(leaky(out1[i] + b1[j], ACT_SLOPE));
}

// final: head-mean + bias + leaky + bbox gather, only for selected rows
__global__ void k_final(const float* __restrict__ out2,
                        const float* __restrict__ b2,
                        const int* __restrict__ bbox,
                        float* __restrict__ out) {
  const int b = blockIdx.x, c = threadIdx.x;       // block = 128 threads
  const long node = bbox[b];
  float v = 0.f;
#pragma unroll
  for (int h = 0; h < NH2; ++h) v += out2[node * NC2 + h * HID + c];
  v = v * (1.f / (float)NH2) + b2[c];
  out[(long)b * HID + c] = leaky(v, ACT_SLOPE);
}

// ---------------------------------------------------------------- driver
extern "C" void kernel_launch(void* const* d_in, const int* in_sizes, int n_in,
                              void* d_out, int out_size, void* d_ws, size_t ws_size,
                              hipStream_t stream) {
  (void)in_sizes; (void)n_in; (void)out_size; (void)ws_size;
  const float* x    = (const float*)d_in[0];
  const int*   ei   = (const int*)d_in[1];
  const int*   bbox = (const int*)d_in[2];
  const float* W1   = (const float*)d_in[3];
  const float* as1  = (const float*)d_in[4];
  const float* ad1  = (const float*)d_in[5];
  const float* b1   = (const float*)d_in[6];
  const float* W2   = (const float*)d_in[7];
  const float* as2  = (const float*)d_in[8];
  const float* ad2  = (const float*)d_in[9];
  const float* b2   = (const float*)d_in[10];
  float* out = (float*)d_out;

  char* ws = (char*)d_ws;
  size_t off = 0;
  auto carve = [&](size_t bytes) -> void* {
    void* p = ws + off;
    off = (off + bytes + 255) & ~(size_t)255;
    return p;
  };
  unsigned short* XBF  = (unsigned short*)carve((size_t)NN * KIN * 2);
  unsigned short* W1T  = (unsigned short*)carve((size_t)NC1 * KIN * 2);
  unsigned short* W2T  = (unsigned short*)carve((size_t)NC2 * NC1 * 2);
  float*          HW   = (float*)carve((size_t)NN * NC1 * 4);   // gemm out (L1), reused L2
  float*          OUT1 = (float*)carve((size_t)NN * NC1 * 4);   // aggr out (L1), reused L2
  unsigned short* H1BF = (unsigned short*)carve((size_t)NN * NC1 * 2);
  float*          AS   = (float*)carve((size_t)NN * NH1 * 4);
  float*          AD   = (float*)carve((size_t)NN * NH1 * 4);
  unsigned int*   SMAX = (unsigned int*)carve((size_t)NN * NH1 * 4);
  float*          SSUM = (float*)carve((size_t)NN * NH1 * 4);

  const int B = 256;
  auto cdiv = [](long a, long b) { return (int)((a + b - 1) / b); };

  // ---- prep: bf16 copies / transposed weights
  k_f32_to_bf16<<<cdiv((long)NN * KIN, B), B, 0, stream>>>(x, XBF, (long)NN * KIN);
  k_transpose_bf16<<<dim3(cdiv(KIN, B), NC1), B, 0, stream>>>(W1, W1T, KIN, NC1);
  k_transpose_bf16<<<dim3(cdiv(NC1, B), NC2), B, 0, stream>>>(W2, W2T, NC1, NC2);

  // ================= layer 1 (5 heads, concat) =================
  k_wmma_gemm<<<dim3(NN / 16, NC1 / 128), B, 0, stream>>>(XBF, W1T, HW, NN, KIN, NC1);
  k_att_scores<<<cdiv((long)NN * NH1, 8), B, 0, stream>>>(HW, as1, ad1, AS, AD, NH1);

  k_zero_u32<<<cdiv((long)NN * NH1, B), B, 0, stream>>>(SMAX, (long)NN * NH1);
  k_zero_u32<<<cdiv((long)NN * NH1, B), B, 0, stream>>>((unsigned int*)SSUM, (long)NN * NH1);
  k_zero_u32<<<cdiv((long)NN * NC1, B), B, 0, stream>>>((unsigned int*)OUT1, (long)NN * NC1);

  k_edge_max <<<dim3(cdiv(ET, B), NH1), B, 0, stream>>>(ei, AS, AD, SMAX, NH1);
  k_edge_sum <<<dim3(cdiv(ET, B), NH1), B, 0, stream>>>(ei, AS, AD, SMAX, SSUM, NH1);
  k_edge_aggr<<<dim3(cdiv(ET, 8), NH1), B, 0, stream>>>(ei, AS, AD, SMAX, SSUM, HW, OUT1, NH1);

  k_post1<<<cdiv((long)NN * NC1, B), B, 0, stream>>>(OUT1, b1, H1BF);

  // ================= layer 2 (3 heads, mean) ===================
  k_wmma_gemm<<<dim3(NN / 16, NC2 / 128), B, 0, stream>>>(H1BF, W2T, HW, NN, NC1, NC2);
  k_att_scores<<<cdiv((long)NN * NH2, 8), B, 0, stream>>>(HW, as2, ad2, AS, AD, NH2);

  k_zero_u32<<<cdiv((long)NN * NH2, B), B, 0, stream>>>(SMAX, (long)NN * NH2);
  k_zero_u32<<<cdiv((long)NN * NH2, B), B, 0, stream>>>((unsigned int*)SSUM, (long)NN * NH2);
  k_zero_u32<<<cdiv((long)NN * NC2, B), B, 0, stream>>>((unsigned int*)OUT1, (long)NN * NC2);

  k_edge_max <<<dim3(cdiv(ET, B), NH2), B, 0, stream>>>(ei, AS, AD, SMAX, NH2);
  k_edge_sum <<<dim3(cdiv(ET, B), NH2), B, 0, stream>>>(ei, AS, AD, SMAX, SSUM, NH2);
  k_edge_aggr<<<dim3(cdiv(ET, 8), NH2), B, 0, stream>>>(ei, AS, AD, SMAX, SSUM, HW, OUT1, NH2);

  k_final<<<NBB, HID, 0, stream>>>(OUT1, b2, bbox, out);
}
